// TransformerBlock_80805514707417
// MI455X (gfx1250) — matrix operand order repaired and checked by reading, not compiled, not run on hardware
//
#include <hip/hip_runtime.h>
#include <hip/hip_bf16.h>
#include <math.h>

// ---------------------------------------------------------------------------
// TransformerBlock on gfx1250: bf16 WMMA (v_wmma_f32_16x16x32_bf16) for all
// matmuls, f32 for LN / softmax / GELU / residuals. Banded attention
// (|i-j|<=128): only a 320-wide key window per 64-query block is computed;
// the rest of the 256MB attn output is exactly 0 (f32 exp underflow) and is
// memset once. GEMM K-loop is register double-buffered so VMEM overlaps WMMA.
// ---------------------------------------------------------------------------

#define S_LEN   2048
#define D_MODEL 1024
#define N_HEADS 16
#define DEPTH   64
#define FF_SIZE 4096
#define LN_EPS  1e-5f

typedef __bf16 v16bf __attribute__((ext_vector_type(16)));
typedef __bf16 v8bf  __attribute__((ext_vector_type(8)));
typedef float  v8f   __attribute__((ext_vector_type(8)));

// ---- WMMA fragment loaders (wave32 layouts per CDNA5 ISA 7.12.2) ----------
// A (16x32 bf16): lane<16 holds K = k0+{0..7}, k0+{16..23}; lane>=16 holds
// K = k0+{8..15}, k0+{24..31}. Caller passes p = row_base + k0 + half*8.
__device__ inline v16bf ld_frag_a(const __bf16* p) {
  v8bf lo = *(const v8bf*)(p);
  v8bf hi = *(const v8bf*)(p + 16);
  v16bf r;
#pragma unroll
  for (int i = 0; i < 8; ++i) { r[i] = lo[i]; r[i + 8] = hi[i]; }
  return r;
}
// B (32x16 bf16): lane holds column n = lane&15, K = k0+half*8+{0..7},
// k0+16+half*8+{0..7}. Caller passes p = col_base + k0 + half*8 (B stored K-contiguous
// per output column, i.e. B^T row-major).
__device__ inline v16bf ld_frag_b(const __bf16* p) {
  v8bf lo = *(const v8bf*)(p);
  v8bf hi = *(const v8bf*)(p + 16);
  v16bf r;
#pragma unroll
  for (int i = 0; i < 8; ++i) { r[i] = lo[i]; r[i + 8] = hi[i]; }
  return r;
}

// ---- weight convert + transpose: W[K,N] f32 -> WT[N,K] bf16 (LDS tiled) ---
__global__ __launch_bounds__(256) void cvt_transpose_kernel(
    const float* __restrict__ W, __bf16* __restrict__ WT, int K, int N) {
  __shared__ float tile[32][33];
  int tx = threadIdx.x, ty = threadIdx.y;
  int n0 = blockIdx.x * 32, k0 = blockIdx.y * 32;
#pragma unroll
  for (int i = 0; i < 4; ++i)
    tile[ty + 8 * i][tx] = W[(size_t)(k0 + ty + 8 * i) * N + n0 + tx];
  __syncthreads();
#pragma unroll
  for (int i = 0; i < 4; ++i)
    WT[(size_t)(n0 + ty + 8 * i) * K + k0 + tx] = (__bf16)tile[tx][ty + 8 * i];
}

// ---- V transpose (LDS tiled): v[s, c] bf16 -> vT[c, s] bf16 ---------------
__global__ __launch_bounds__(256) void transpose_v_kernel(
    const __bf16* __restrict__ v, __bf16* __restrict__ vT) {
  __shared__ __bf16 tile[32][33];
  int tx = threadIdx.x, ty = threadIdx.y;
  int c0 = blockIdx.x * 32, s0 = blockIdx.y * 32;
#pragma unroll
  for (int i = 0; i < 4; ++i)
    tile[ty + 8 * i][tx] = v[(size_t)(s0 + ty + 8 * i) * D_MODEL + c0 + tx];
  __syncthreads();
#pragma unroll
  for (int i = 0; i < 4; ++i)
    vT[(size_t)(c0 + ty + 8 * i) * S_LEN + s0 + tx] = tile[tx][ty + 8 * i];
}

// ---- LayerNorm (f32 in, bf16 out) -----------------------------------------
__global__ __launch_bounds__(256) void layernorm_kernel(
    const float* __restrict__ x, const float* __restrict__ g,
    const float* __restrict__ b, __bf16* __restrict__ out) {
  __shared__ float red[256];
  int row = blockIdx.x, tid = threadIdx.x;
  const float* xr = x + (size_t)row * D_MODEL;
  float vals[4], s = 0.f;
#pragma unroll
  for (int i = 0; i < 4; ++i) { vals[i] = xr[tid + i * 256]; s += vals[i]; }
  red[tid] = s; __syncthreads();
  for (int off = 128; off > 0; off >>= 1) {
    if (tid < off) red[tid] += red[tid + off];
    __syncthreads();
  }
  float mu = red[0] * (1.0f / D_MODEL);
  __syncthreads();
  float sq = 0.f;
#pragma unroll
  for (int i = 0; i < 4; ++i) { float d = vals[i] - mu; sq += d * d; }
  red[tid] = sq; __syncthreads();
  for (int off = 128; off > 0; off >>= 1) {
    if (tid < off) red[tid] += red[tid + off];
    __syncthreads();
  }
  float rstd = rsqrtf(red[0] * (1.0f / D_MODEL) + LN_EPS);
#pragma unroll
  for (int i = 0; i < 4; ++i) {
    int c = tid + i * 256;
    out[(size_t)row * D_MODEL + c] = (__bf16)((vals[i] - mu) * rstd * g[c] + b[c]);
  }
}

// ---- generic bf16 GEMM: C[M,N] = act(A[M,K] @ BT[N,K]^T + bias) (+res) ----
// 256 threads = 8 waves; block tile 128x128; wave tile 16x128; K step 64,
// register double-buffered (loads for next 32-half issue before WMMA batch).
__global__ __launch_bounds__(256) void gemm_bf16_kernel(
    const __bf16* __restrict__ A, const __bf16* __restrict__ BT,
    const float* __restrict__ bias, const float* __restrict__ res,
    float* __restrict__ outf, __bf16* __restrict__ outh,
    int M, int N, int K, int act) {
  int lane = threadIdx.x & 31;
  int wave = threadIdx.x >> 5;
  int half = lane >> 4;
  int lm   = lane & 15;
  int m0 = blockIdx.y * 128 + wave * 16;
  int n0 = blockIdx.x * 128;

  v8f zero = {};
  v8f acc[8];
#pragma unroll
  for (int t = 0; t < 8; ++t) acc[t] = zero;

  const __bf16* arow = A + (size_t)(m0 + lm) * K + half * 8;
  const __bf16* brow[8];
#pragma unroll
  for (int t = 0; t < 8; ++t)
    brow[t] = BT + (size_t)(n0 + t * 16 + lm) * K + half * 8;

  v16bf a0, a1, b0[8], b1[8];
  a0 = ld_frag_a(arow);
#pragma unroll
  for (int t = 0; t < 8; ++t) b0[t] = ld_frag_b(brow[t]);

  for (int kk = 0; kk < K; kk += 64) {              // K % 64 == 0 always
    // prefetch second 32-half of this chunk
    a1 = ld_frag_a(arow + kk + 32);
#pragma unroll
    for (int t = 0; t < 8; ++t) b1[t] = ld_frag_b(brow[t] + kk + 32);
    // compute on first half while second half is in flight
#pragma unroll
    for (int t = 0; t < 8; ++t)
      acc[t] = __builtin_amdgcn_wmma_f32_16x16x32_bf16(
          false, a0, false, b0[t], (short)0, acc[t], false, false);
    // prefetch first half of next chunk
    if (kk + 64 < K) {
      a0 = ld_frag_a(arow + kk + 64);
#pragma unroll
      for (int t = 0; t < 8; ++t) b0[t] = ld_frag_b(brow[t] + kk + 64);
    }
    // compute on second half
#pragma unroll
    for (int t = 0; t < 8; ++t)
      acc[t] = __builtin_amdgcn_wmma_f32_16x16x32_bf16(
          false, a1, false, b1[t], (short)0, acc[t], false, false);
  }

#pragma unroll
  for (int t = 0; t < 8; ++t) {
    int col = n0 + t * 16 + lm;
    float bv = bias[col];
#pragma unroll
    for (int j = 0; j < 8; ++j) {
      int row = m0 + j + half * 8;
      float v = acc[t][j] + bv;
      if (act) v = 0.5f * v * (1.0f + erff(v * 0.70710678118654752f)); // exact GELU
      size_t idx = (size_t)row * N + col;
      if (res)  v += res[idx];
      if (outf) outf[idx] = v;
      if (outh) outh[idx] = (__bf16)v;
    }
  }
}

// ---- banded attention -----------------------------------------------------
// grid (S/64, H), 128 threads = 4 waves; each WG: 64 query rows, key window
// [r0-128, r0+192) = 320 cols. LDS: 64x320 f32 scores + 64x320 bf16 probs.
#define WIN 320
__global__ __launch_bounds__(128) void attention_kernel(
    const __bf16* __restrict__ qh, const __bf16* __restrict__ kh,
    const __bf16* __restrict__ vT, float* __restrict__ attn_out,
    __bf16* __restrict__ out_h) {
  extern __shared__ char smem[];
  float*  sc = (float*)smem;                        // [64][320]
  __bf16* pb = (__bf16*)(smem + 64 * WIN * 4);      // [64][320]
  __shared__ float s_part[128];

  int head = blockIdx.y;
  int r0   = blockIdx.x * 64;
  int c0   = r0 - 128;                              // global col of window col 0
  int lane = threadIdx.x & 31;
  int wave = threadIdx.x >> 5;
  int half = lane >> 4;
  int lm   = lane & 15;
  const float NEG_INF = -__builtin_huge_valf();

  // ---- phase 1: scores = Q K^T / 8, masked ----
  int wr = wave * 16;                               // wave's local row base
  const __bf16* qrow = qh + (size_t)(r0 + wr + lm) * D_MODEL + head * DEPTH;
  v16bf aq0 = ld_frag_a(qrow + half * 8);           // invariant across tiles
  v16bf aq1 = ld_frag_a(qrow + 32 + half * 8);
  for (int ct = 0; ct < WIN / 16; ++ct) {           // uniform per block
    int tcol = c0 + ct * 16;
    if (tcol < 0 || tcol + 16 > S_LEN) {            // window tile outside seq
#pragma unroll
      for (int j = 0; j < 8; ++j)
        sc[(wr + j + half * 8) * WIN + ct * 16 + lm] = NEG_INF;
      continue;
    }
    const __bf16* krow = kh + (size_t)(tcol + lm) * D_MODEL + head * DEPTH + half * 8;
    v16bf bk0 = ld_frag_b(krow);
    v16bf bk1 = ld_frag_b(krow + 32);
    v8f acc = {};
    acc = __builtin_amdgcn_wmma_f32_16x16x32_bf16(
        false, aq0, false, bk0, (short)0, acc, false, false);
    acc = __builtin_amdgcn_wmma_f32_16x16x32_bf16(
        false, aq1, false, bk1, (short)0, acc, false, false);
#pragma unroll
    for (int j = 0; j < 8; ++j) {
      int i  = r0 + wr + j + half * 8;
      int jc = tcol + lm;
      float s = acc[j] * 0.125f;                    // 1/sqrt(64)
      int dd = i - jc; if (dd < 0) dd = -dd;
      if (dd > 128) s = -10000.0f;                  // matches reference mask
      sc[(wr + j + half * 8) * WIN + ct * 16 + lm] = s;
    }
  }
  __syncthreads();

  // ---- phase 2: row softmax (2 threads/row); band -> global f32,
  //      probs -> LDS bf16 ----
  {
    int lr = threadIdx.x & 63;                      // row
    int hp = threadIdx.x >> 6;                      // half 0/1 of the row
    int i  = r0 + lr;
    float* row = sc + (size_t)lr * WIN;
    int cbeg = hp * (WIN / 2), cend = cbeg + WIN / 2;
    float mx = NEG_INF;
    for (int c = cbeg; c < cend; ++c) mx = fmaxf(mx, row[c]);
    s_part[threadIdx.x] = mx; __syncthreads();
    mx = fmaxf(s_part[lr], s_part[lr + 64]); __syncthreads();
    float sum = 0.f;
    for (int c = cbeg; c < cend; ++c) {
      float e = expf(row[c] - mx); row[c] = e; sum += e;
    }
    s_part[threadIdx.x] = sum; __syncthreads();
    sum = s_part[lr] + s_part[lr + 64];
    float inv = 1.0f / sum;
    float* arow = attn_out + ((size_t)head * S_LEN + i) * S_LEN;
    for (int c = cbeg; c < cend; ++c) {
      float p = row[c] * inv;
      pb[(size_t)lr * WIN + c] = (__bf16)p;
      int jc = c0 + c;
      int dd = i - jc; if (dd < 0) dd = -dd;
      if (jc >= 0 && jc < S_LEN && dd <= 128) arow[jc] = p;
    }
  }
  __syncthreads();

  // ---- phase 3: out = probs @ V (vT[c, s] -> contiguous B frags) ----------
  v8f zero = {};
  v8f oacc[4];
#pragma unroll
  for (int dt = 0; dt < 4; ++dt) oacc[dt] = zero;
  const __bf16* prow = pb + (size_t)(wr + lm) * WIN + half * 8;
  for (int kt = 0; kt < WIN / 16; kt += 2) {        // K step 32, uniform
    int tg = c0 + kt * 16;
    if (tg < 0 || tg + 32 > S_LEN) continue;        // probs there are 0 anyway
    v16bf a = ld_frag_a(prow + kt * 16);
    v16bf bv[4];
#pragma unroll
    for (int dt = 0; dt < 4; ++dt)
      bv[dt] = ld_frag_b(vT + ((size_t)head * DEPTH + dt * 16 + lm) * S_LEN +
                         tg + half * 8);
#pragma unroll
    for (int dt = 0; dt < 4; ++dt)
      oacc[dt] = __builtin_amdgcn_wmma_f32_16x16x32_bf16(
          false, a, false, bv[dt], (short)0, oacc[dt], false, false);
  }
#pragma unroll
  for (int dt = 0; dt < 4; ++dt) {
#pragma unroll
    for (int j = 0; j < 8; ++j) {
      int row = r0 + wr + j + half * 8;
      int col = head * DEPTH + dt * 16 + lm;
      out_h[(size_t)row * D_MODEL + col] = (__bf16)oacc[dt][j];
    }
  }
}

// ---------------------------------------------------------------------------
extern "C" void kernel_launch(void* const* d_in, const int* in_sizes, int n_in,
                              void* d_out, int out_size, void* d_ws, size_t ws_size,
                              hipStream_t stream) {
  (void)in_sizes; (void)n_in; (void)out_size; (void)ws_size;
  const float* x     = (const float*)d_in[0];
  const float* wq_w  = (const float*)d_in[1];
  const float* wq_b  = (const float*)d_in[2];
  const float* wk_w  = (const float*)d_in[3];
  const float* wk_b  = (const float*)d_in[4];
  const float* wv_w  = (const float*)d_in[5];
  const float* wv_b  = (const float*)d_in[6];
  const float* fc_w  = (const float*)d_in[7];
  const float* fc_b  = (const float*)d_in[8];
  const float* ff1_w = (const float*)d_in[9];
  const float* ff1_b = (const float*)d_in[10];
  const float* ff2_w = (const float*)d_in[11];
  const float* ff2_b = (const float*)d_in[12];
  const float* ln1_g = (const float*)d_in[13];
  const float* ln1_b = (const float*)d_in[14];
  const float* ln2_g = (const float*)d_in[15];
  const float* ln2_b = (const float*)d_in[16];

  float* out_x    = (float*)d_out;                       // [S, D] f32
  float* out_attn = out_x + (size_t)S_LEN * D_MODEL;     // [H, S, S] f32

  // workspace carve-up
  char* ws = (char*)d_ws;
  size_t off = 0;
  auto alloc = [&](size_t bytes) {
    char* p = ws + off;
    off += (bytes + 255) & ~(size_t)255;
    return p;
  };
  __bf16* wqT  = (__bf16*)alloc((size_t)D_MODEL * D_MODEL * 2);
  __bf16* wkT  = (__bf16*)alloc((size_t)D_MODEL * D_MODEL * 2);
  __bf16* wvT  = (__bf16*)alloc((size_t)D_MODEL * D_MODEL * 2);
  __bf16* fcT  = (__bf16*)alloc((size_t)D_MODEL * D_MODEL * 2);
  __bf16* ff1T = (__bf16*)alloc((size_t)D_MODEL * FF_SIZE * 2);
  __bf16* ff2T = (__bf16*)alloc((size_t)FF_SIZE * D_MODEL * 2);
  __bf16* xn1h = (__bf16*)alloc((size_t)S_LEN * D_MODEL * 2);
  __bf16* qh   = (__bf16*)alloc((size_t)S_LEN * D_MODEL * 2);
  __bf16* kh   = (__bf16*)alloc((size_t)S_LEN * D_MODEL * 2);
  __bf16* vh   = (__bf16*)alloc((size_t)S_LEN * D_MODEL * 2);
  __bf16* vTh  = (__bf16*)alloc((size_t)S_LEN * D_MODEL * 2);
  __bf16* aoh  = (__bf16*)alloc((size_t)S_LEN * D_MODEL * 2);
  float*  x1   = (float*) alloc((size_t)S_LEN * D_MODEL * 4);
  __bf16* xn2h = (__bf16*)alloc((size_t)S_LEN * D_MODEL * 2);
  __bf16* f1h  = (__bf16*)alloc((size_t)S_LEN * FF_SIZE * 2);

  // zero the (mostly-zero) 256MB banded attention output
  hipMemsetAsync(out_attn, 0, (size_t)N_HEADS * S_LEN * S_LEN * sizeof(float), stream);

  // weights -> bf16 transposed [N,K] (tiled transpose, coalesced both sides)
  auto cvt = [&](const float* W, __bf16* WT, int K, int N) {
    cvt_transpose_kernel<<<dim3(N / 32, K / 32), dim3(32, 8), 0, stream>>>(W, WT, K, N);
  };
  cvt(wq_w,  wqT,  D_MODEL, D_MODEL);
  cvt(wk_w,  wkT,  D_MODEL, D_MODEL);
  cvt(wv_w,  wvT,  D_MODEL, D_MODEL);
  cvt(fc_w,  fcT,  D_MODEL, D_MODEL);
  cvt(ff1_w, ff1T, D_MODEL, FF_SIZE);
  cvt(ff2_w, ff2T, FF_SIZE, D_MODEL);

  // LN1
  layernorm_kernel<<<S_LEN, 256, 0, stream>>>(x, ln1_g, ln1_b, xn1h);

  // Q, K, V projections (bf16 out)
  dim3 gP(D_MODEL / 128, S_LEN / 128);
  gemm_bf16_kernel<<<gP, 256, 0, stream>>>(xn1h, wqT, wq_b, nullptr, nullptr, qh,
                                           S_LEN, D_MODEL, D_MODEL, 0);
  gemm_bf16_kernel<<<gP, 256, 0, stream>>>(xn1h, wkT, wk_b, nullptr, nullptr, kh,
                                           S_LEN, D_MODEL, D_MODEL, 0);
  gemm_bf16_kernel<<<gP, 256, 0, stream>>>(xn1h, wvT, wv_b, nullptr, nullptr, vh,
                                           S_LEN, D_MODEL, D_MODEL, 0);
  transpose_v_kernel<<<dim3(D_MODEL / 32, S_LEN / 32), dim3(32, 8), 0, stream>>>(vh, vTh);

  // banded attention: writes attn f32 + context bf16
  size_t lds_bytes = (size_t)64 * WIN * (4 + 2);   // 122880 B < 320KB/WGP
  attention_kernel<<<dim3(S_LEN / 64, N_HEADS), 128, lds_bytes, stream>>>(
      qh, kh, vTh, out_attn, aoh);

  // fc projection + residual: x1 = x + attn_ctx @ fc + b
  gemm_bf16_kernel<<<gP, 256, 0, stream>>>(aoh, fcT, fc_b, x, x1, nullptr,
                                           S_LEN, D_MODEL, D_MODEL, 0);
  // LN2
  layernorm_kernel<<<S_LEN, 256, 0, stream>>>(x1, ln2_g, ln2_b, xn2h);

  // FFN: gelu(xn2 @ ff1 + b1) @ ff2 + b2, + x1 -> final output
  gemm_bf16_kernel<<<dim3(FF_SIZE / 128, S_LEN / 128), 256, 0, stream>>>(
      xn2h, ff1T, ff1_b, nullptr, nullptr, f1h, S_LEN, FF_SIZE, D_MODEL, 1);
  gemm_bf16_kernel<<<gP, 256, 0, stream>>>(f1h, ff2T, ff2_b, x1, out_x, nullptr,
                                           S_LEN, D_MODEL, FF_SIZE, 0);
}
